// LocalConv_87076166959108
// MI455X (gfx1250) — matrix operand order repaired and checked by reading, not compiled, not run on hardware
//
#include <hip/hip_runtime.h>

// Problem constants (from reference)
#define B_    64
#define C_    16
#define H_    64
#define W_    64
#define OH_   62
#define OW_   62
#define P_    (OH_ * OW_)     // 3844
#define FEAT_ 144             // C*KH*KW
#define OC_   32
#define KPAD  168             // padded LDS K row (f16 elems): 16B-aligned frags, de-banked
#define KTILES 5              // ceil(160/32) K-steps of 32

typedef __attribute__((ext_vector_type(16))) _Float16 v16h;
typedef __attribute__((ext_vector_type(8)))  _Float16 v8h;
typedef __attribute__((ext_vector_type(8)))  float    v8f;

__global__ __launch_bounds__(256)
void localconv_wmma_f16(const float* __restrict__ in,
                        const float* __restrict__ ker,
                        float* __restrict__ out)
{
    __shared__ __align__(32) _Float16 lA[B_  * KPAD];  // patches: [b][k]  (64 x 168)
    __shared__ __align__(32) _Float16 lB[OC_ * KPAD];  // kernel transposed: [o][k] (32 x 168)

    const int p   = blockIdx.x;
    const int oh  = p / OW_;
    const int ow  = p % OW_;
    const int tid = threadIdx.x;

    // ---- stage kernel[p] (FEAT x OC, f32 row-major) -> lB[o][f] f16 (transposed)
    {
        const float4* kp4 = (const float4*)(ker + (size_t)p * (FEAT_ * OC_));
        for (int i = tid; i < (FEAT_ * OC_) / 4; i += 256) {  // 1152 float4, coalesced
            float4 v = kp4[i];
            int base = i * 4;
            int f = base / OC_;
            int o = base % OC_;           // OC_%4==0 -> o..o+3 share f
            lB[(o + 0) * KPAD + f] = (_Float16)v.x;
            lB[(o + 1) * KPAD + f] = (_Float16)v.y;
            lB[(o + 2) * KPAD + f] = (_Float16)v.z;
            lB[(o + 3) * KPAD + f] = (_Float16)v.w;
        }
        for (int i = tid; i < OC_ * (160 - FEAT_); i += 256) {  // zero-pad K [144,160)
            int o = i / 16, f = FEAT_ + (i % 16);
            lB[o * KPAD + f] = (_Float16)0.f;
        }
    }

    // ---- gather patches -> lA[b][f] f16, f = c*9 + ki*3 + kj
    for (int i = tid; i < B_ * FEAT_; i += 256) {
        int b = i / FEAT_, f = i % FEAT_;
        int c = f / 9, r = f % 9, ki = r / 3, kj = r % 3;
        float v = in[(((size_t)b * C_ + c) * H_ + (oh + ki)) * W_ + (ow + kj)];
        lA[b * KPAD + f] = (_Float16)v;
    }
    for (int i = tid; i < B_ * (160 - FEAT_); i += 256) {       // zero-pad K [144,160)
        int b = i / 16, f = FEAT_ + (i % 16);
        lA[b * KPAD + f] = (_Float16)0.f;
    }

    __syncthreads();

    // ---- per-wave 16x16 output tile, K-loop of v_wmma_f32_16x16x32_f16
    const int wave = tid >> 5;
    const int lane = tid & 31;
    const int mt   = wave & 3;    // batch tile (M)
    const int nt   = wave >> 2;   // out-channel tile (N)
    const int half = lane >> 4;
    const int l16  = lane & 15;

    const _Float16* Arow = &lA[(mt * 16 + l16) * KPAD];  // A: lane holds row M=l16 (both halves)
    const _Float16* Brow = &lB[(nt * 16 + l16) * KPAD];  // B: lane holds col N=l16

    v8f acc = {};
#pragma unroll
    for (int k0 = 0; k0 < KTILES; ++k0) {
        // A fragment (16x32 f16): lane gets contiguous K = 32k0 + 16*half + e
        int ka = k0 * 32 + 16 * half;
        v8h a0 = *(const v8h*)(Arow + ka);
        v8h a1 = *(const v8h*)(Arow + ka + 8);
        v16h a = __builtin_shufflevector(a0, a1, 0, 1, 2, 3, 4, 5, 6, 7,
                                                 8, 9, 10, 11, 12, 13, 14, 15);
        // B fragment (32x16 f16): lane gets contiguous K = 32k0 + 16*half + e
        int kb = k0 * 32 + 16 * half;
        v16h b = *(const v16h*)(Brow + kb);

        acc = __builtin_amdgcn_wmma_f32_16x16x32_f16(
            /*neg_a=*/false, a, /*neg_b=*/false, b,
            /*c_mod=*/(short)0, acc, /*reuse_a=*/false, /*reuse_b=*/false);
    }

    // ---- store: C/D layout -> VGPR r: M = 8*half + r, N = l16
    const int bbase = mt * 16 + 8 * half;
    const int o     = nt * 16 + l16;
#pragma unroll
    for (int r = 0; r < 8; ++r) {
        out[((size_t)(bbase + r) * OC_ + o) * P_ + p] = acc[r];
    }
}

extern "C" void kernel_launch(void* const* d_in, const int* in_sizes, int n_in,
                              void* d_out, int out_size, void* d_ws, size_t ws_size,
                              hipStream_t stream)
{
    const float* in  = (const float*)d_in[0];   // (B, C, H, W) f32
    const float* ker = (const float*)d_in[1];   // (P, FEAT, OC) f32
    float* out = (float*)d_out;                 // (B, OC, OH, OW) f32

    (void)in_sizes; (void)n_in; (void)out_size; (void)d_ws; (void)ws_size;

    localconv_wmma_f16<<<P_, 256, 0, stream>>>(in, ker, out);
}
